// TransFusionHeadAnchorMatching_52501680226714
// MI455X (gfx1250) — matrix operand order repaired and checked by reading, not compiled, not run on hardware
//
#include <hip/hip_runtime.h>
#include <math.h>

typedef __attribute__((ext_vector_type(16))) _Float16 v16h;
typedef __attribute__((ext_vector_type(8)))  float    v8f;

#define HH 180
#define WW 180
#define HWSZ 32400
#define PW 182
#define PSZ (182*182)
#define BB 2
#define CINX 512
#define CH 128
#define NPROP 200
#define NCLS 10

__constant__ float c_anchors[10][3] = {
 {4.63f,1.97f,1.74f},{6.93f,2.51f,2.84f},{6.37f,2.85f,3.19f},
 {10.5f,2.94f,3.47f},{12.29f,2.9f,3.87f},{0.5f,2.53f,0.98f},
 {2.11f,0.77f,1.47f},{1.7f,0.6f,1.28f},{0.73f,0.67f,1.77f},
 {0.41f,0.41f,1.07f}};

__device__ __forceinline__ float clip_sig(float x){
  float s = 1.f/(1.f+__expf(-x));
  return fminf(fmaxf(s,1e-4f),1.f-1e-4f);
}

// ---------------- input f32 -> padded f16 ----------------
__global__ void k_pack_input(const float* __restrict__ in, _Float16* __restrict__ xpad, int total){
  int tid = blockIdx.x*blockDim.x + threadIdx.x; if(tid>=total) return;
  int x = tid % WW; int y = (tid/WW) % HH; int c = tid / HWSZ;   // c = b*CINX+cin
  xpad[(size_t)c*PSZ + (size_t)(y+1)*PW + (x+1)] = (_Float16)in[tid];
}

// -------- pack conv weights [Cout][Cin][3][3] into WMMA A-fragments --------
// K ordering: k = kidx*Cin + cin, kidx = ky*3+kx. Cin power of two.
__global__ void k_pack_wconv(const float* __restrict__ w, _Float16* __restrict__ frag,
                             int Cin, int cl, int KS, int total){
  int tid = blockIdx.x*blockDim.x + threadIdx.x; if(tid>=total) return;
  int e = tid & 15; int t2 = tid >> 4; int lane = t2 & 31; int t3 = t2 >> 5;
  int ks = t3 % KS; int ct = t3 / KS;
  int hi = lane >> 4, m = lane & 15;
  int klocal = hi*16 + e;
  int k = ks*32 + klocal;
  int cin = k & (Cin-1); int kidx = k >> cl;
  frag[tid] = (_Float16) w[((size_t)(ct*16+m)*Cin + cin)*9 + kidx];
}

// -------- pack dense GEMM weights [M][K] into WMMA A-fragments --------
__global__ void k_pack_wgemm(const float* __restrict__ w, _Float16* __restrict__ frag,
                             int K, int KS, int total){
  int tid = blockIdx.x*blockDim.x + threadIdx.x; if(tid>=total) return;
  int e = tid & 15; int t2 = tid >> 4; int lane = t2 & 31; int t3 = t2 >> 5;
  int ks = t3 % KS; int mt = t3 / KS;
  int hi = lane >> 4, m = lane & 15;
  int klocal = hi*16 + e;
  int k = ks*32 + klocal;
  frag[tid] = (_Float16) w[(size_t)(mt*16+m)*K + k];
}

// -------- implicit-GEMM 3x3 conv via V_WMMA_F32_16X16X32_F16 --------
// grid (HWSZ/16, Cout/16, B), block 32 (one wave, EXEC all ones)
__global__ void __launch_bounds__(32)
k_conv3x3_wmma(const _Float16* __restrict__ inpad, const _Float16* __restrict__ wfrag,
               const float* __restrict__ gamma, const float* __restrict__ beta, int relu,
               float* __restrict__ outf32, _Float16* __restrict__ outpad,
               int Cin, int cl, int KS, int Cout){
  int lane = threadIdx.x; int tile = blockIdx.x; int ct = blockIdx.y; int b = blockIdx.z;
  int hi = lane >> 4, lo = lane & 15;
  int p = tile*16 + lo; int oy = p/WW, ox = p%WW;
  const _Float16* inb = inpad + (size_t)b*Cin*PSZ;
  v8f acc = {0.f,0.f,0.f,0.f,0.f,0.f,0.f,0.f};
  for(int ks=0; ks<KS; ++ks){
    v16h a = *(const v16h*)(wfrag + (((size_t)ct*KS + ks)*32 + lane)*16);
    int kb = ks*32 + hi*16;
    v16h bf;
#pragma unroll
    for(int e=0; e<16; ++e){
      int k = kb + e;
      int cin = k & (Cin-1);
      int kidx = k >> cl;
      int ky = kidx/3, kx = kidx - ky*3;
      bf[e] = inb[(size_t)cin*PSZ + (size_t)(oy+ky)*PW + (ox+kx)];
    }
    acc = __builtin_amdgcn_wmma_f32_16x16x32_f16(false, a, false, bf, (short)0, acc, false, false);
  }
#pragma unroll
  for(int r=0; r<8; ++r){
    int co = ct*16 + r + hi*8;
    float v = acc[r];
    if (gamma) v = v*gamma[co] + beta[co]; else v += beta[co];
    if (relu) v = fmaxf(v, 0.f);
    if (outf32) outf32[((size_t)b*Cout + co)*HWSZ + p] = v;
    if (outpad) outpad[((size_t)b*Cout + co)*PSZ + (size_t)(oy+1)*PW + (ox+1)] = (_Float16)v;
  }
}

// -------- dense GEMM: out[b][n][m] = bmat[b][n][:] . wfrag_row(m) + bias[m] --------
// bmat f16 [B][N][K] row-major, grid (N/16, M/16, B), block 32
__global__ void __launch_bounds__(32)
k_gemm_wmma(const _Float16* __restrict__ bmat, const _Float16* __restrict__ afrag,
            const float* __restrict__ bias, float* __restrict__ out,
            int N, int K, int M){
  int lane = threadIdx.x; int nt = blockIdx.x; int mt = blockIdx.y; int b = blockIdx.z;
  int hi = lane >> 4, lo = lane & 15;
  int KS = K >> 5;
  const _Float16* brow = bmat + ((size_t)b*N + (size_t)nt*16 + lo)*K;
  v8f acc = {0.f,0.f,0.f,0.f,0.f,0.f,0.f,0.f};
  for(int ks=0; ks<KS; ++ks){
    v16h a = *(const v16h*)(afrag + (((size_t)mt*KS + ks)*32 + lane)*16);
    v16h bf = *(const v16h*)(brow + ks*32 + hi*16);
    acc = __builtin_amdgcn_wmma_f32_16x16x32_f16(false, a, false, bf, (short)0, acc, false, false);
  }
  int gp = nt*16 + lo;
#pragma unroll
  for(int r=0; r<8; ++r){
    int gm = mt*16 + r + hi*8;
    out[((size_t)b*N + gp)*M + gm] = acc[r] + bias[gm];
  }
}

// -------- small-Cout 3x3 conv (Cout=1 or 3) from padded f16 --------
__global__ void k_conv2_small(const _Float16* __restrict__ mid, const float* __restrict__ w2,
                              const float* __restrict__ b2, float* __restrict__ out,
                              int cout, int total){
  int tid = blockIdx.x*blockDim.x + threadIdx.x; if(tid>=total) return;
  int co = tid % cout; int p = (tid/cout) % HWSZ; int b = tid/(cout*HWSZ);
  int oy = p/WW, ox = p%WW;
  const _Float16* base = mid + (size_t)b*CH*PSZ + (size_t)oy*PW + ox;
  float s = b2[co];
  for(int cin=0; cin<CH; ++cin){
    const _Float16* ip = base + (size_t)cin*PSZ;
    const float* wr = w2 + ((size_t)co*CH + cin)*9;
    s += (float)ip[0]*wr[0] + (float)ip[1]*wr[1] + (float)ip[2]*wr[2]
       + (float)ip[PW]*wr[3] + (float)ip[PW+1]*wr[4] + (float)ip[PW+2]*wr[5]
       + (float)ip[2*PW]*wr[6] + (float)ip[2*PW+1]*wr[7] + (float)ip[2*PW+2]*wr[8];
  }
  out[tid] = s;  // [b][p][cout]
}

// -------- anchor text vectors, normalized --------
__global__ void k_anchor_prep(const float* __restrict__ qvals, float* __restrict__ anchor_n){
  int a = blockIdx.x; int t = threadIdx.x;
  __shared__ float sv[90];
  __shared__ float sinv;
  if (t < 90){
    float v = c_anchors[a][t/30] - qvals[t%30];
    v = fmaxf(v, 0.f);
    sv[t] = v/(v + 1e-5f);
  }
  __syncthreads();
  if (t == 0){
    float s = 0.f;
    for(int i=0;i<90;++i) s += sv[i]*sv[i];
    sinv = 1.f/(1e-5f + sqrtf(s));
  }
  __syncthreads();
  if (t < 90) anchor_n[a*90 + t] = sv[t]*sinv;
}

// -------- fused dense heatmap: dv, normalize, cosine sim, sigmoid * ag --------
__global__ void k_heatmap(const float* __restrict__ dh, const float* __restrict__ agraw,
                          const float* __restrict__ anchor_n, const float* __restrict__ qvals,
                          const float* __restrict__ lsp, const float* __restrict__ lbp,
                          float* __restrict__ hmout, int total){
  __shared__ float san[900];
  __shared__ float sq[30];
  for(int i=threadIdx.x; i<900; i+=blockDim.x) san[i] = anchor_n[i];
  if (threadIdx.x < 30) sq[threadIdx.x] = qvals[threadIdx.x];
  __syncthreads();
  int tid = blockIdx.x*blockDim.x + threadIdx.x; if(tid>=total) return;
  int b = tid / HWSZ; int p = tid % HWSZ;
  float d3[3]; d3[0]=dh[(size_t)tid*3]; d3[1]=dh[(size_t)tid*3+1]; d3[2]=dh[(size_t)tid*3+2];
  float accv[10];
#pragma unroll
  for(int k=0;k<10;++k) accv[k]=0.f;
  float nrm = 0.f;
  for(int dm=0; dm<3; ++dm){
    float xv = d3[dm];
    for(int bn=0; bn<30; ++bn){
      float v = fmaxf(xv - sq[bn], 0.f);
      float u = v/(v + 1e-5f);
      nrm += u*u;
      int f = dm*30 + bn;
#pragma unroll
      for(int k=0;k<10;++k) accv[k] += u*san[k*90 + f];
    }
  }
  float inv = 1.f/(1e-5f + sqrtf(nrm));
  float ls = lsp[0], lb = lbp[0];
  float agv = clip_sig(agraw[tid]);
#pragma unroll
  for(int k=0;k<10;++k){
    hmout[((size_t)b*NCLS + k)*HWSZ + p] = clip_sig(ls*accv[k]*inv + lb)*agv;
  }
}

// -------- 3x3 local-max NMS --------
__global__ void k_nms(const float* __restrict__ hm, float* __restrict__ nmsb, int total){
  int tid = blockIdx.x*blockDim.x + threadIdx.x; if(tid>=total) return;
  int p = tid % HWSZ; int k = (tid/HWSZ) % NCLS; int b = tid/(NCLS*HWSZ);
  const float* hmc = hm + ((size_t)b*NCLS + k)*HWSZ;
  float v = hmc[p];
  int y = p/WW, x = p%WW;
  float lm;
  if (k >= 8) lm = v;
  else if (y>=1 && y<=HH-2 && x>=1 && x<=WW-2){
    lm = -1e30f;
    for(int dy=-1; dy<=1; ++dy)
      for(int dx=-1; dx<=1; ++dx)
        lm = fmaxf(lm, hmc[(y+dy)*WW + (x+dx)]);
  } else lm = 0.f;
  nmsb[tid] = (v == lm) ? v : 0.f;
}

// -------- iterative top-200 (stable, lowest index wins ties), destroys buf --------
__global__ void k_topk(float* __restrict__ buf, int* __restrict__ posidx, int* __restrict__ clsidx){
  int b = blockIdx.x; int t = threadIdx.x;
  const int NTH = 1024; const int NTOT = NCLS*HWSZ;
  __shared__ float sv[NTH];
  __shared__ int   si[NTH];
  float* a = buf + (size_t)b*NTOT;
  for(int it=0; it<NPROP; ++it){
    float bv = -1e30f; int bi = 0x7fffffff;
    for(int i=t; i<NTOT; i+=NTH){
      float v = a[i];
      if (v > bv){ bv = v; bi = i; }
    }
    sv[t]=bv; si[t]=bi; __syncthreads();
    for(int s=NTH/2; s>0; s>>=1){
      if (t < s){
        float v2 = sv[t+s]; int i2 = si[t+s];
        if (v2 > sv[t] || (v2 == sv[t] && i2 < si[t])){ sv[t]=v2; si[t]=i2; }
      }
      __syncthreads();
    }
    if (t == 0){
      int best = si[0];
      posidx[b*NPROP + it] = best % HWSZ;
      clsidx[b*NPROP + it] = best / HWSZ;
      a[best] = -1.f;
    }
    __syncthreads();
  }
}

__global__ void k_qpos(const int* __restrict__ posidx, float* __restrict__ qpos, int total){
  int tid = blockIdx.x*blockDim.x + threadIdx.x; if(tid>=total) return;
  int pos = posidx[tid];
  qpos[(size_t)tid*2]   = (float)(pos/WW) + 0.5f;
  qpos[(size_t)tid*2+1] = (float)(pos%WW) + 0.5f;
}

// -------- gather query features + anchor encoding --------
__global__ void k_gather_enc(const float* __restrict__ lidar, const int* __restrict__ posidx,
                             const int* __restrict__ clsidx, const float* __restrict__ anchor_n,
                             const float* __restrict__ encw, const float* __restrict__ encb,
                             float* __restrict__ x0, int total){
  int tid = blockIdx.x*blockDim.x + threadIdx.x; if(tid>=total) return;
  int c = tid & 127; int qq = tid >> 7;   // qq = b*NPROP+q
  int pos = posidx[qq]; int cls = clsidx[qq];
  int b = qq / NPROP;
  const float* av = anchor_n + cls*90;
  const float* wr = encw + (size_t)c*90;
  float s = encb[c];
  for(int f=0; f<90; ++f) s += wr[f]*av[f];
  x0[tid] = lidar[((size_t)b*CH + c)*HWSZ + pos] + s;
}

// -------- conv1d(2->C)+BN+relu+conv1d(C->C), block per point --------
__global__ void k_posemb(const float* __restrict__ xy, const float* __restrict__ w1,
                         const float* __restrict__ b1, const float* __restrict__ g,
                         const float* __restrict__ bt, const float* __restrict__ w2,
                         const float* __restrict__ b2, float* __restrict__ out){
  int n = blockIdx.x; int t = threadIdx.x;
  __shared__ float sh[CH];
  float x0 = xy[(size_t)n*2], x1 = xy[(size_t)n*2+1];
  float h = w1[t*2]*x0 + w1[t*2+1]*x1 + b1[t];
  h = fmaxf(h*g[t] + bt[t], 0.f);
  sh[t] = h; __syncthreads();
  float o = b2[t];
  const float* wr = w2 + (size_t)t*CH;
  for(int c=0;c<CH;++c) o += wr[c]*sh[c];
  out[(size_t)n*CH + t] = o;
}

// -------- kmem + cross pos-emb -> f16, block per bev position --------
__global__ void k_kfull(const float* __restrict__ lidar, const float* __restrict__ w1,
                        const float* __restrict__ b1, const float* __restrict__ g,
                        const float* __restrict__ bt, const float* __restrict__ w2,
                        const float* __restrict__ b2, _Float16* __restrict__ kfull16){
  int n = blockIdx.x; int t = threadIdx.x;
  int b = n / HWSZ; int p = n % HWSZ;
  __shared__ float sh[CH];
  float x0 = (float)(p/WW) + 0.5f, x1 = (float)(p%WW) + 0.5f;
  float h = w1[t*2]*x0 + w1[t*2+1]*x1 + b1[t];
  h = fmaxf(h*g[t] + bt[t], 0.f);
  sh[t] = h; __syncthreads();
  float o = b2[t];
  const float* wr = w2 + (size_t)t*CH;
  for(int c=0;c<CH;++c) o += wr[c]*sh[c];
  o += lidar[((size_t)b*CH + t)*HWSZ + p];
  kfull16[(size_t)n*CH + t] = (_Float16)o;
}

// -------- generic linear: out[r][o] = in[r][:] . w[o][:] + bias[o] --------
__global__ void k_linear(const float* __restrict__ in, const float* __restrict__ w,
                         const float* __restrict__ bias, float* __restrict__ out,
                         int rows, int K, int O, int relu){
  int tid = blockIdx.x*blockDim.x + threadIdx.x; if(tid >= rows*O) return;
  int o = tid % O; int r = tid / O;
  const float* ir = in + (size_t)r*K;
  const float* wr = w + (size_t)o*K;
  float s = bias[o];
  for(int k=0;k<K;++k) s += ir[k]*wr[k];
  if (relu) s = fmaxf(s, 0.f);
  out[(size_t)r*O + o] = s;
}

__global__ void k_add(const float* __restrict__ a, const float* __restrict__ b,
                      float* __restrict__ o, int total){
  int tid = blockIdx.x*blockDim.x + threadIdx.x; if(tid>=total) return;
  o[tid] = a[tid] + b[tid];
}

// -------- residual add + layernorm, block(128) per row --------
__global__ void k_add_ln(const float* __restrict__ x, const float* __restrict__ y,
                         const float* __restrict__ g, const float* __restrict__ bp,
                         float* __restrict__ out){
  int r = blockIdx.x; int t = threadIdx.x;
  __shared__ float red[CH];
  float v = x[(size_t)r*CH + t] + y[(size_t)r*CH + t];
  red[t] = v; __syncthreads();
  for(int s=64; s>0; s>>=1){ if(t<s) red[t]+=red[t+s]; __syncthreads(); }
  float m = red[0]*(1.f/128.f); __syncthreads();
  float d = v - m;
  red[t] = d*d; __syncthreads();
  for(int s=64; s>0; s>>=1){ if(t<s) red[t]+=red[t+s]; __syncthreads(); }
  float var = red[0]*(1.f/128.f);
  out[(size_t)r*CH + t] = d*rsqrtf(var + 1e-5f)*g[t] + bp[t];
}

// -------- self-attn scores + softmax, block per (b,h,q) --------
__global__ void k_sa_softmax(const float* __restrict__ qkv, float* __restrict__ prob){
  int qi = blockIdx.x, h = blockIdx.y, b = blockIdx.z, ki = threadIdx.x;
  __shared__ float red[256];
  float sc = -1e30f;
  if (ki < NPROP){
    const float* qr = qkv + ((size_t)(b*NPROP + qi))*384 + h*16;
    const float* kr = qkv + ((size_t)(b*NPROP + ki))*384 + 128 + h*16;
    float s = 0.f;
#pragma unroll
    for(int d=0; d<16; ++d) s += qr[d]*kr[d];
    sc = s*0.25f;
  }
  red[ki] = sc; __syncthreads();
  for(int s=128; s>0; s>>=1){ if(ki<s) red[ki]=fmaxf(red[ki],red[ki+s]); __syncthreads(); }
  float mx = red[0]; __syncthreads();
  float e = (ki < NPROP) ? __expf(sc - mx) : 0.f;
  red[ki] = e; __syncthreads();
  for(int s=128; s>0; s>>=1){ if(ki<s) red[ki]+=red[ki+s]; __syncthreads(); }
  float sm = red[0];
  if (ki < NPROP) prob[(((size_t)(b*8 + h))*NPROP + qi)*NPROP + ki] = e/sm;
}

__global__ void k_sa_av(const float* __restrict__ prob, const float* __restrict__ qkv,
                        float* __restrict__ attno, int total){
  int tid = blockIdx.x*blockDim.x + threadIdx.x; if(tid>=total) return;
  int d = tid & 15; int h = (tid>>4) & 7; int q = (tid>>7) % NPROP; int b = tid/(NPROP*128);
  const float* pr = prob + (((size_t)(b*8 + h))*NPROP + q)*NPROP;
  float s = 0.f;
  for(int ki=0; ki<NPROP; ++ki)
    s += pr[ki]*qkv[((size_t)(b*NPROP + ki))*384 + 256 + h*16 + d];
  attno[((size_t)(b*NPROP + q))*CH + h*16 + d] = s;
}

// -------- cross-attn, flash-style, one wave per (b,h,q) over 32400 keys --------
__global__ void __launch_bounds__(32)
k_cross_attn(const float* __restrict__ qproj, const float* __restrict__ kvproj,
             float* __restrict__ attno){
  int qi = blockIdx.x, h = blockIdx.y, b = blockIdx.z, lane = threadIdx.x;
  float qv[16];
  const float* qb = qproj + ((size_t)(b*NPROP + qi))*CH + h*16;
#pragma unroll
  for(int d=0; d<16; ++d) qv[d] = qb[d];
  float m = -1e30f, s = 0.f;
  float acc[16];
#pragma unroll
  for(int d=0; d<16; ++d) acc[d] = 0.f;
  for(int kk=lane; kk<HWSZ; kk+=32){
    const float* kp = kvproj + ((size_t)b*HWSZ + kk)*256 + h*16;
    float sc = 0.f;
#pragma unroll
    for(int d=0; d<16; ++d) sc += qv[d]*kp[d];
    sc *= 0.25f;
    float nm = fmaxf(m, sc);
    float sca = __expf(m - nm);
    float e = __expf(sc - nm);
    s = s*sca + e;
#pragma unroll
    for(int d=0; d<16; ++d) acc[d] = acc[d]*sca + e*kp[128 + d];
    m = nm;
  }
  __shared__ float shm[32], shs[32], sha[32][16];
  shm[lane] = m; shs[lane] = s;
#pragma unroll
  for(int d=0; d<16; ++d) sha[lane][d] = acc[d];
  __syncthreads();
  if (lane < 16){
    float M = -1e30f;
    for(int l=0; l<32; ++l) M = fmaxf(M, shm[l]);
    float S = 0.f, A = 0.f;
    for(int l=0; l<32; ++l){
      float w = (shs[l] > 0.f) ? __expf(shm[l] - M) : 0.f;
      S += shs[l]*w;
      A += sha[l][lane]*w;
    }
    attno[((size_t)(b*NPROP + qi))*CH + h*16 + lane] = A/S;
  }
}

// -------- prediction head: relu(w1.x * g1 + b1) -> w2 + b2, write into preds --------
__global__ void k_head(const float* __restrict__ x3, const float* __restrict__ w1,
                       const float* __restrict__ g1, const float* __restrict__ b1,
                       const float* __restrict__ w2, const float* __restrict__ b2,
                       int cout, int ch0, const float* __restrict__ qpos,
                       float* __restrict__ preds){
  int r = blockIdx.x; int t = threadIdx.x;
  int b = r / NPROP; int q = r % NPROP;
  __shared__ float sx[CH], shh[CH];
  sx[t] = x3[(size_t)r*CH + t]; __syncthreads();
  float s = 0.f;
  const float* wr = w1 + (size_t)t*CH;
  for(int c=0; c<CH; ++c) s += wr[c]*sx[c];
  s = fmaxf(s*g1[t] + b1[t], 0.f);
  shh[t] = s; __syncthreads();
  if (t < cout){
    float o = b2[t];
    const float* w2r = w2 + (size_t)t*CH;
    for(int c=0; c<CH; ++c) o += w2r[c]*shh[c];
    if (qpos) o += qpos[(size_t)r*2 + t];
    preds[((size_t)b*13 + ch0 + t)*NPROP + q] = o;
  }
}

__global__ void k_qhm(const float* __restrict__ hm, const int* __restrict__ posidx,
                      float* __restrict__ qhm, int total){
  int tid = blockIdx.x*blockDim.x + threadIdx.x; if(tid>=total) return;
  int q = tid % NPROP; int k = (tid/NPROP) % NCLS; int b = tid/(NPROP*NCLS);
  qhm[tid] = hm[((size_t)b*NCLS + k)*HWSZ + posidx[b*NPROP + q]];
}

// ---------------- host side ----------------
extern "C" void kernel_launch(void* const* d_in, const int* in_sizes, int n_in,
                              void* d_out, int out_size, void* d_ws, size_t ws_size,
                              hipStream_t stream){
  (void)in_sizes; (void)n_in; (void)out_size; (void)ws_size;
  // jax pytree flattening order (dict keys sorted alphabetically at each level)
  const float* p_inputs   = (const float*)d_in[0];
  const float* ag_b1 = (const float*)d_in[1];  const float* ag_b2 = (const float*)d_in[2];
  const float* ag_g1 = (const float*)d_in[3];  const float* ag_w1 = (const float*)d_in[4];
  const float* ag_w2 = (const float*)d_in[5];
  const float* enc_b = (const float*)d_in[6];  const float* enc_w = (const float*)d_in[7];
  const float* ca_bin = (const float*)d_in[8]; const float* ca_bout = (const float*)d_in[9];
  const float* ca_win = (const float*)d_in[10];const float* ca_wout = (const float*)d_in[11];
  const float* cpe_b1 = (const float*)d_in[12];const float* cpe_b2 = (const float*)d_in[13];
  const float* cpe_bt = (const float*)d_in[14];const float* cpe_g  = (const float*)d_in[15];
  const float* cpe_w1 = (const float*)d_in[16];const float* cpe_w2 = (const float*)d_in[17];
  const float* lin1_b = (const float*)d_in[18];const float* lin1_w = (const float*)d_in[19];
  const float* lin2_b = (const float*)d_in[20];const float* lin2_w = (const float*)d_in[21];
  const float* ln1_b = (const float*)d_in[22]; const float* ln1_g = (const float*)d_in[23];
  const float* ln2_b = (const float*)d_in[24]; const float* ln2_g = (const float*)d_in[25];
  const float* ln3_b = (const float*)d_in[26]; const float* ln3_g = (const float*)d_in[27];
  const float* sa_bin = (const float*)d_in[28];const float* sa_bout = (const float*)d_in[29];
  const float* sa_win = (const float*)d_in[30];const float* sa_wout = (const float*)d_in[31];
  const float* spe_b1 = (const float*)d_in[32];const float* spe_b2 = (const float*)d_in[33];
  const float* spe_bt = (const float*)d_in[34];const float* spe_g  = (const float*)d_in[35];
  const float* spe_w1 = (const float*)d_in[36];const float* spe_w2 = (const float*)d_in[37];
  // heads sorted: center, dim, heatmap, height, rot, vel ; each b1,b2,g1,w1,w2
  const float* hc_b1=(const float*)d_in[38]; const float* hc_b2=(const float*)d_in[39];
  const float* hc_g1=(const float*)d_in[40]; const float* hc_w1=(const float*)d_in[41];
  const float* hc_w2=(const float*)d_in[42];
  const float* hd_b1=(const float*)d_in[43]; const float* hd_b2=(const float*)d_in[44];
  const float* hd_g1=(const float*)d_in[45]; const float* hd_w1=(const float*)d_in[46];
  const float* hd_w2=(const float*)d_in[47];
  const float* hh_b1=(const float*)d_in[48]; const float* hh_b2=(const float*)d_in[49];
  const float* hh_g1=(const float*)d_in[50]; const float* hh_w1=(const float*)d_in[51];
  const float* hh_w2=(const float*)d_in[52];
  const float* ht_b1=(const float*)d_in[53]; const float* ht_b2=(const float*)d_in[54];
  const float* ht_g1=(const float*)d_in[55]; const float* ht_w1=(const float*)d_in[56];
  const float* ht_w2=(const float*)d_in[57];
  const float* hr_b1=(const float*)d_in[58]; const float* hr_b2=(const float*)d_in[59];
  const float* hr_g1=(const float*)d_in[60]; const float* hr_w1=(const float*)d_in[61];
  const float* hr_w2=(const float*)d_in[62];
  const float* hv_b1=(const float*)d_in[63]; const float* hv_b2=(const float*)d_in[64];
  const float* hv_g1=(const float*)d_in[65]; const float* hv_w1=(const float*)d_in[66];
  const float* hv_w2=(const float*)d_in[67];
  const float* hm_b1=(const float*)d_in[68]; const float* hm_b2=(const float*)d_in[69];
  const float* hm_g1=(const float*)d_in[70]; const float* hm_w1=(const float*)d_in[71];
  const float* hm_w2=(const float*)d_in[72];
  const float* logit_b = (const float*)d_in[73];
  const float* logit_s = (const float*)d_in[74];
  const float* qvals   = (const float*)d_in[75];
  const float* sh_b = (const float*)d_in[76];
  const float* sh_w = (const float*)d_in[77];

  char* ws = (char*)d_ws;
  size_t off = 0;
  auto WSA = [&](size_t bytes){ size_t o = off; off += (bytes + 255) & ~(size_t)255; return o; };
  const size_t szXPAD = (size_t)BB*CINX*PSZ*2;         // 67.8 MB (reused as KVPROJ later)
  size_t oXPAD   = WSA(szXPAD);
  size_t oFEAT   = WSA((size_t)BB*CH*PSZ*2);
  size_t oMID    = WSA((size_t)BB*CH*PSZ*2);
  size_t oWFSH   = WSA((size_t)CH*CINX*9*2);
  size_t oWFHD   = WSA((size_t)CH*CH*9*2);
  size_t oWFKV   = WSA((size_t)256*CH*2);
  size_t oLIDAR  = WSA((size_t)BB*CH*HWSZ*4);
  size_t oDH     = WSA((size_t)BB*HWSZ*3*4);
  size_t oAG     = WSA((size_t)BB*HWSZ*4);
  size_t oANCH   = WSA(900*4);
  size_t oNMS    = WSA((size_t)BB*NCLS*HWSZ*4);
  size_t oPOSI   = WSA((size_t)BB*NPROP*4);
  size_t oCLSI   = WSA((size_t)BB*NPROP*4);
  size_t oQPOS   = WSA((size_t)BB*NPROP*2*4);
  size_t oKF16   = WSA((size_t)BB*HWSZ*CH*2);
  size_t oX0     = WSA((size_t)BB*NPROP*CH*4);
  size_t oQPE    = WSA((size_t)BB*NPROP*CH*4);
  size_t oXQ     = WSA((size_t)BB*NPROP*CH*4);
  size_t oSQKV   = WSA((size_t)BB*NPROP*384*4);
  size_t oPROB   = WSA((size_t)BB*8*NPROP*NPROP*4);
  size_t oATTN   = WSA((size_t)BB*NPROP*CH*4);
  size_t oTMP    = WSA((size_t)BB*NPROP*CH*4);
  size_t oX1     = WSA((size_t)BB*NPROP*CH*4);
  size_t oQPROJ  = WSA((size_t)BB*NPROP*CH*4);
  size_t oX2     = WSA((size_t)BB*NPROP*CH*4);
  size_t oF1     = WSA((size_t)BB*NPROP*256*4);
  size_t oX3     = WSA((size_t)BB*NPROP*CH*4);

  _Float16* xpad = (_Float16*)(ws + oXPAD);
  _Float16* feat = (_Float16*)(ws + oFEAT);
  _Float16* mid  = (_Float16*)(ws + oMID);
  _Float16* wfsh = (_Float16*)(ws + oWFSH);
  _Float16* wfhd = (_Float16*)(ws + oWFHD);
  _Float16* wfkv = (_Float16*)(ws + oWFKV);
  float* lidar = (float*)(ws + oLIDAR);
  float* dhb   = (float*)(ws + oDH);
  float* agb   = (float*)(ws + oAG);
  float* anch  = (float*)(ws + oANCH);
  float* nmsb  = (float*)(ws + oNMS);
  int* posi    = (int*)(ws + oPOSI);
  int* clsi    = (int*)(ws + oCLSI);
  float* qpos  = (float*)(ws + oQPOS);
  _Float16* kf16 = (_Float16*)(ws + oKF16);
  float* x0 = (float*)(ws + oX0);   float* qpe = (float*)(ws + oQPE);
  float* xq = (float*)(ws + oXQ);   float* sqkv = (float*)(ws + oSQKV);
  float* prob = (float*)(ws + oPROB); float* attn = (float*)(ws + oATTN);
  float* tmp = (float*)(ws + oTMP); float* x1 = (float*)(ws + oX1);
  float* qproj = (float*)(ws + oQPROJ); float* x2 = (float*)(ws + oX2);
  float* f1 = (float*)(ws + oF1);   float* x3 = (float*)(ws + oX3);
  float* kvproj = (float*)(ws + oXPAD);   // reuse XPAD region after shared conv

  float* out_preds = (float*)d_out;
  float* out_hm    = (float*)d_out + (size_t)BB*13*NPROP;
  float* out_qhm   = out_hm + (size_t)BB*NCLS*HWSZ;

  auto GB = [](long n, int b){ return dim3((unsigned)((n + b - 1)/b)); };

  // zero padded buffers (borders must be zero)
  hipMemsetAsync(xpad, 0, szXPAD, stream);
  hipMemsetAsync(feat, 0, (size_t)BB*CH*PSZ*2, stream);
  hipMemsetAsync(mid,  0, (size_t)BB*CH*PSZ*2, stream);

  // input -> padded f16
  long npi = (long)BB*CINX*HWSZ;
  k_pack_input<<<GB(npi,256),256,0,stream>>>(p_inputs, xpad, (int)npi);

  // shared conv (512->128), WMMA
  int KSsh = CINX*9/32;
  long npw = (long)(CH/16)*KSsh*512;
  k_pack_wconv<<<GB(npw,256),256,0,stream>>>(sh_w, wfsh, CINX, 9, KSsh, (int)npw);
  k_conv3x3_wmma<<<dim3(HWSZ/16, CH/16, BB), 32, 0, stream>>>(
      xpad, wfsh, nullptr, sh_b, 0, lidar, feat, CINX, 9, KSsh, CH);

  // hm head conv1 (128->128) WMMA, then small conv2 (->3)
  int KShd = CH*9/32;
  long npw2 = (long)(CH/16)*KShd*512;
  k_pack_wconv<<<GB(npw2,256),256,0,stream>>>(hm_w1, wfhd, CH, 7, KShd, (int)npw2);
  k_conv3x3_wmma<<<dim3(HWSZ/16, CH/16, BB), 32, 0, stream>>>(
      feat, wfhd, hm_g1, hm_b1, 1, nullptr, mid, CH, 7, KShd, CH);
  long nc2 = (long)BB*HWSZ*3;
  k_conv2_small<<<GB(nc2,256),256,0,stream>>>(mid, hm_w2, hm_b2, dhb, 3, (int)nc2);

  // ag head conv1 (reuse mid buffer) then conv2 (->1)
  k_pack_wconv<<<GB(npw2,256),256,0,stream>>>(ag_w1, wfhd, CH, 7, KShd, (int)npw2);
  k_conv3x3_wmma<<<dim3(HWSZ/16, CH/16, BB), 32, 0, stream>>>(
      feat, wfhd, ag_g1, ag_b1, 1, nullptr, mid, CH, 7, KShd, CH);
  long nc1 = (long)BB*HWSZ;
  k_conv2_small<<<GB(nc1,256),256,0,stream>>>(mid, ag_w2, ag_b2, agb, 1, (int)nc1);

  // anchors + dense heatmap
  k_anchor_prep<<<10, 96, 0, stream>>>(qvals, anch);
  long nhp = (long)BB*HWSZ;
  k_heatmap<<<GB(nhp,128),128,0,stream>>>(dhb, agb, anch, qvals, logit_s, logit_b, out_hm, (int)nhp);

  // NMS + top-k
  long nnm = (long)BB*NCLS*HWSZ;
  k_nms<<<GB(nnm,256),256,0,stream>>>(out_hm, nmsb, (int)nnm);
  k_topk<<<BB, 1024, 0, stream>>>(nmsb, posi, clsi);

  // gather queries
  k_qpos<<<GB(BB*NPROP,256),256,0,stream>>>(posi, qpos, BB*NPROP);
  long nge = (long)BB*NPROP*CH;
  k_gather_enc<<<GB(nge,256),256,0,stream>>>(lidar, posi, clsi, anch, enc_w, enc_b, x0, (int)nge);

  // positional embeddings
  k_posemb<<<BB*NPROP, CH, 0, stream>>>(qpos, spe_w1, spe_b1, spe_g, spe_bt, spe_w2, spe_b2, qpe);
  k_kfull<<<BB*HWSZ, CH, 0, stream>>>(lidar, cpe_w1, cpe_b1, cpe_g, cpe_bt, cpe_w2, cpe_b2, kf16);

  // cross-attn K/V projection via WMMA GEMM: [B,32400,128] x [256,128]^T
  long npkv = (long)16*4*512;
  k_pack_wgemm<<<GB(npkv,256),256,0,stream>>>(ca_win + 128*128, wfkv, CH, 4, (int)npkv);
  k_gemm_wmma<<<dim3(HWSZ/16, 256/16, BB), 32, 0, stream>>>(kf16, wfkv, ca_bin + 128, kvproj, HWSZ, CH, 256);

  // self-attention
  int nrow = BB*NPROP;
  k_add<<<GB((long)nrow*CH,256),256,0,stream>>>(x0, qpe, xq, nrow*CH);
  k_linear<<<GB((long)nrow*384,256),256,0,stream>>>(xq, sa_win, sa_bin, sqkv, nrow, CH, 384, 0);
  k_sa_softmax<<<dim3(NPROP,8,BB), 256, 0, stream>>>(sqkv, prob);
  k_sa_av<<<GB((long)nrow*CH,256),256,0,stream>>>(prob, sqkv, attn, nrow*CH);
  k_linear<<<GB((long)nrow*CH,256),256,0,stream>>>(attn, sa_wout, sa_bout, tmp, nrow, CH, CH, 0);
  k_add_ln<<<nrow, CH, 0, stream>>>(x0, tmp, ln1_g, ln1_b, x1);

  // cross-attention
  k_add<<<GB((long)nrow*CH,256),256,0,stream>>>(x1, qpe, xq, nrow*CH);
  k_linear<<<GB((long)nrow*CH,256),256,0,stream>>>(xq, ca_win, ca_bin, qproj, nrow, CH, CH, 0);
  k_cross_attn<<<dim3(NPROP,8,BB), 32, 0, stream>>>(qproj, kvproj, attn);
  k_linear<<<GB((long)nrow*CH,256),256,0,stream>>>(attn, ca_wout, ca_bout, tmp, nrow, CH, CH, 0);
  k_add_ln<<<nrow, CH, 0, stream>>>(x1, tmp, ln2_g, ln2_b, x2);

  // FFN
  k_linear<<<GB((long)nrow*256,256),256,0,stream>>>(x2, lin1_w, lin1_b, f1, nrow, CH, 256, 1);
  k_linear<<<GB((long)nrow*CH,256),256,0,stream>>>(f1, lin2_w, lin2_b, tmp, nrow, 256, CH, 0);
  k_add_ln<<<nrow, CH, 0, stream>>>(x2, tmp, ln3_g, ln3_b, x3);

  // prediction heads -> preds [B,13,200]: center(0) height(2) dim(3) rot(6) vel(8) heatmap(10)
  k_head<<<nrow, CH, 0, stream>>>(x3, hc_w1, hc_g1, hc_b1, hc_w2, hc_b2, 2, 0,  qpos,    out_preds);
  k_head<<<nrow, CH, 0, stream>>>(x3, ht_w1, ht_g1, ht_b1, ht_w2, ht_b2, 1, 2,  nullptr, out_preds);
  k_head<<<nrow, CH, 0, stream>>>(x3, hd_w1, hd_g1, hd_b1, hd_w2, hd_b2, 3, 3,  nullptr, out_preds);
  k_head<<<nrow, CH, 0, stream>>>(x3, hr_w1, hr_g1, hr_b1, hr_w2, hr_b2, 2, 6,  nullptr, out_preds);
  k_head<<<nrow, CH, 0, stream>>>(x3, hv_w1, hv_g1, hv_b1, hv_w2, hv_b2, 2, 8,  nullptr, out_preds);
  k_head<<<nrow, CH, 0, stream>>>(x3, hh_w1, hh_g1, hh_b1, hh_w2, hh_b2, 3, 10, nullptr, out_preds);

  // q_hm gather
  k_qhm<<<GB((long)BB*NCLS*NPROP,256),256,0,stream>>>(out_hm, posi, out_qhm, BB*NCLS*NPROP);
}
